// BiAttentionLayer_25967372272330
// MI455X (gfx1250) — matrix operand order repaired and checked by reading, not compiled, not run on hardware
//
#include <hip/hip_runtime.h>

#define B_  8
#define TC  4096
#define TQ  1024
#define DD  256
#define M_TILE 64
#define JC  128

#define CH_STRIDE 264   // halves, padded (264*2 = 528 B, 16B multiple)
#define Q_STRIDE  264
#define QT_STRIDE 136
#define P_STRIDE  136

#define OFF_QRM   0
#define OFF_QT    (OFF_QRM + JC*Q_STRIDE*2)        // 67584
#define OFF_CH    (OFF_QT  + DD*QT_STRIDE*2)       // 137216
#define OFF_P     (OFF_CH  + M_TILE*CH_STRIDE*2)   // 171008
#define OFF_SQ    (OFF_P   + M_TILE*P_STRIDE*2)    // 188416
#define OFF_MW    (OFF_SQ  + TQ*4)                 // 192512
#define OFF_LW    (OFF_MW  + 2*M_TILE*4)
#define OFF_MF    (OFF_LW  + 2*M_TILE*4)
#define OFF_LINV  (OFF_MF  + M_TILE*4)
#define LDS_BYTES (OFF_LINV + M_TILE*4)            // ~194 KB < 320 KB/WGP

typedef __attribute__((ext_vector_type(16))) _Float16 v16h;
typedef __attribute__((ext_vector_type(8)))  _Float16 v8h;
typedef __attribute__((ext_vector_type(4)))  _Float16 v4h;
typedef __attribute__((ext_vector_type(8)))  float    v8f;

__device__ __forceinline__ v16h pack16(v8h lo, v8h hi) {
  v16h r;
#pragma unroll
  for (int i = 0; i < 8; ++i) { r[i] = lo[i]; r[i + 8] = hi[i]; }
  return r;
}

// ---------------- kernel 0: zero h accumulator, compute s_q = Q . w2 ----------------
__global__ void bidaf_setup(const float* __restrict__ qst, const float* __restrict__ w,
                            float* __restrict__ h_acc, float* __restrict__ sq_g) {
  int t = blockIdx.x * 256 + threadIdx.x;
  if (t < B_ * DD) h_acc[t] = 0.0f;
  if (t < B_ * TQ) {
    const float* q  = qst + (size_t)t * DD;
    const float* w2 = w + DD;
    float s = 0.0f;
#pragma unroll 4
    for (int d = 0; d < DD; d += 4) {
      float4 a  = *(const float4*)(q  + d);
      float4 bb = *(const float4*)(w2 + d);
      s += a.x * bb.x + a.y * bb.y + a.z * bb.z + a.w * bb.w;
    }
    sq_g[t] = s;
  }
}

// ---------------- kernel 1: fused scores -> softmax -> U, plus h partial sums --------
__global__ __launch_bounds__(256)
void bidaf_main(const float* __restrict__ ctx, const float* __restrict__ qst,
                const float* __restrict__ w, const float* __restrict__ sq_g,
                float* __restrict__ h_acc, float* __restrict__ out) {
  extern __shared__ char smem[];
  _Float16* sQrm = (_Float16*)(smem + OFF_QRM);  // Q chunk, row-major  (K=d contig)
  _Float16* sQt  = (_Float16*)(smem + OFF_QT);   // Q chunk, transposed (K=j contig)
  _Float16* sCh  = (_Float16*)(smem + OFF_CH);   // (C * w3) tile, f16
  _Float16* sP   = (_Float16*)(smem + OFF_P);    // probability tile, f16
  float* sSq   = (float*)(smem + OFF_SQ);
  float* sMw   = (float*)(smem + OFF_MW);
  float* sLw   = (float*)(smem + OFF_LW);
  float* sMf   = (float*)(smem + OFF_MF);
  float* sLinv = (float*)(smem + OFF_LINV);

  const int tid  = threadIdx.x;
  const int wv   = tid >> 5;
  const int lane = tid & 31;
  const int n = lane & 15, g = lane >> 4;
  const int rb  = wv >> 1;          // row block 0..3 (16 rows each)
  const int ct0 = (wv & 1) * 4;     // first of 4 score col-tiles for this wave

  const int b  = blockIdx.x / (TC / M_TILE);
  const int i0 = (blockIdx.x % (TC / M_TILE)) * M_TILE;

  // ---- stage 0: load C-hat = C*w3 (f16) and s_q row for this batch ----
  {
    const float* w3 = w + 2 * DD;
    for (int q = tid; q < M_TILE * (DD / 4); q += 256) {
      int i = q >> 6, dc = (q & 63) << 2;
      float4 c4 = *(const float4*)(ctx + ((size_t)(b * TC) + i0 + i) * DD + dc);
      float4 w4 = *(const float4*)(w3 + dc);
      v4h hc; hc[0] = (_Float16)(c4.x * w4.x); hc[1] = (_Float16)(c4.y * w4.y);
      hc[2] = (_Float16)(c4.z * w4.z); hc[3] = (_Float16)(c4.w * w4.w);
      *(v4h*)(sCh + i * CH_STRIDE + dc) = hc;
    }
    for (int j = tid; j < TQ; j += 256) sSq[j] = sq_g[b * TQ + j];
  }

  auto load_q_chunk = [&](int jbase, bool both) {
    for (int q = tid; q < JC * (DD / 4); q += 256) {
      int j = q >> 6, dc = (q & 63) << 2;
      float4 q4 = *(const float4*)(qst + ((size_t)(b * TQ) + jbase + j) * DD + dc);
      v4h hq; hq[0] = (_Float16)q4.x; hq[1] = (_Float16)q4.y;
      hq[2] = (_Float16)q4.z; hq[3] = (_Float16)q4.w;
      *(v4h*)(sQrm + j * Q_STRIDE + dc) = hq;
      if (both) {
        sQt[(dc + 0) * QT_STRIDE + j] = hq[0];
        sQt[(dc + 1) * QT_STRIDE + j] = hq[1];
        sQt[(dc + 2) * QT_STRIDE + j] = hq[2];
        sQt[(dc + 3) * QT_STRIDE + j] = hq[3];
      }
    }
  };

  // GEMM1: scores tile (4 x 16x16 tiles per wave), K = 256, + s_q bias
  auto gemm1 = [&](int jbase, v8f acc[4]) {
#pragma unroll
    for (int t = 0; t < 4; ++t) acc[t] = {};
    const _Float16* Arow = sCh + (rb * 16) * CH_STRIDE;
#pragma unroll
    for (int k = 0; k < DD; k += 32) {
      const _Float16* ap = Arow + n * CH_STRIDE + k + g * 8;
      v16h a = pack16(*(const v8h*)ap, *(const v8h*)(ap + 16));
#pragma unroll
      for (int t = 0; t < 4; ++t) {
        const _Float16* bp = sQrm + ((ct0 + t) * 16 + n) * Q_STRIDE + k + g * 8;
        v16h bm = pack16(*(const v8h*)bp, *(const v8h*)(bp + 16));
        acc[t] = __builtin_amdgcn_wmma_f32_16x16x32_f16(false, a, false, bm,
                                                        (short)0, acc[t], false, false);
      }
    }
#pragma unroll
    for (int t = 0; t < 4; ++t) {
      float bias = sSq[jbase + (ct0 + t) * 16 + n];
#pragma unroll
      for (int v = 0; v < 8; ++v) acc[t][v] += bias;
    }
  };

  // ---- pass 1: online row max / sum-exp over all of Tq ----
  float m[8], l[8];
#pragma unroll
  for (int v = 0; v < 8; ++v) { m[v] = -3.0e38f; l[v] = 0.0f; }

  for (int c = 0; c < TQ / JC; ++c) {
    __syncthreads();
    load_q_chunk(c * JC, false);
    __syncthreads();
    v8f acc[4];
    gemm1(c * JC, acc);
#pragma unroll
    for (int v = 0; v < 8; ++v) {
      float tm = fmaxf(fmaxf(acc[0][v], acc[1][v]), fmaxf(acc[2][v], acc[3][v]));
      tm = fmaxf(tm, __shfl_xor(tm, 1));
      tm = fmaxf(tm, __shfl_xor(tm, 2));
      tm = fmaxf(tm, __shfl_xor(tm, 4));
      tm = fmaxf(tm, __shfl_xor(tm, 8));
      float nm = fmaxf(m[v], tm);
      float s = __expf(acc[0][v] - nm) + __expf(acc[1][v] - nm) +
                __expf(acc[2][v] - nm) + __expf(acc[3][v] - nm);
      s += __shfl_xor(s, 1); s += __shfl_xor(s, 2);
      s += __shfl_xor(s, 4); s += __shfl_xor(s, 8);
      l[v] = l[v] * __expf(m[v] - nm) + s;
      m[v] = nm;
    }
  }

  // combine the two waves sharing each row block
  if (n == 0) {  // lanes 0 and 16
#pragma unroll
    for (int v = 0; v < 8; ++v) {
      int row = rb * 16 + v + 8 * g;
      sMw[(wv & 1) * M_TILE + row] = m[v];
      sLw[(wv & 1) * M_TILE + row] = l[v];
    }
  }
  __syncthreads();
  if (tid < M_TILE) {
    float m0 = sMw[tid], m1 = sMw[M_TILE + tid];
    float l0 = sLw[tid], l1 = sLw[M_TILE + tid];
    float mfv = fmaxf(m0, m1);
    float lf  = l0 * __expf(m0 - mfv) + l1 * __expf(m1 - mfv);
    sMf[tid]   = mfv;
    sLinv[tid] = 1.0f / lf;   // == b_i = max_j A[i,j]
  }
  __syncthreads();

  // ---- h partial: h[b,d] += sum_i b_i * C[i,d]  (one global f32 atomic per column) ----
  {
    float ha = 0.0f;
#pragma unroll 4
    for (int i = 0; i < M_TILE; ++i)
      ha += sLinv[i] * ctx[((size_t)(b * TC) + i0 + i) * DD + tid];
    atomicAdd(&h_acc[b * DD + tid], ha);
  }

  // ---- pass 2: recompute scores -> P (f16) -> U = P @ Q ----
  float mf[8], linv[8];
#pragma unroll
  for (int v = 0; v < 8; ++v) {
    int row = rb * 16 + v + 8 * g;
    mf[v]   = sMf[row];
    linv[v] = sLinv[row];
  }

  v8f acc2[8];
#pragma unroll
  for (int c = 0; c < 8; ++c) acc2[c] = {};

  for (int c = 0; c < TQ / JC; ++c) {
    __syncthreads();
    load_q_chunk(c * JC, true);
    __syncthreads();
    v8f acc[4];
    gemm1(c * JC, acc);
#pragma unroll
    for (int t = 0; t < 4; ++t) {
#pragma unroll
      for (int v = 0; v < 8; ++v) {
        float p = __expf(acc[t][v] - mf[v]) * linv[v];
        int row = rb * 16 + v + 8 * g;
        int col = (ct0 + t) * 16 + n;
        sP[row * P_STRIDE + col] = (_Float16)p;
      }
    }
    __syncthreads();
    const _Float16* Prow = sP + (rb * 16) * P_STRIDE;
#pragma unroll
    for (int kk = 0; kk < JC; kk += 32) {
      const _Float16* ap = Prow + n * P_STRIDE + kk + g * 8;
      v16h a = pack16(*(const v8h*)ap, *(const v8h*)(ap + 16));
#pragma unroll
      for (int cc = 0; cc < 8; ++cc) {
        int d = ((wv & 1) * 8 + cc) * 16 + n;
        const _Float16* bp = sQt + d * QT_STRIDE + kk + g * 8;
        v16h bm = pack16(*(const v8h*)bp, *(const v8h*)(bp + 16));
        acc2[cc] = __builtin_amdgcn_wmma_f32_16x16x32_f16(false, a, false, bm,
                                                          (short)0, acc2[cc], false, false);
      }
    }
  }

  // ---- epilogue: write G[:, 0:768] = [C, U, C*U] ----
#pragma unroll
  for (int cc = 0; cc < 8; ++cc) {
    int col = ((wv & 1) * 8 + cc) * 16 + n;
#pragma unroll
    for (int v = 0; v < 8; ++v) {
      int row = i0 + rb * 16 + v + 8 * g;
      size_t crow = (size_t)(b * TC) + row;
      float cv = ctx[crow * DD + col];
      float u  = acc2[cc][v];
      float* op = out + crow * (4 * DD);
      op[col]          = cv;
      op[DD + col]     = u;
      op[2 * DD + col] = cv * u;
    }
  }
}

// ---------------- kernel 2: G[:, 768:1024] = C * broadcast(h) ----------------
__global__ void bidaf_tail(const float* __restrict__ ctx, const float* __restrict__ h,
                           float* __restrict__ out) {
  size_t idx = (size_t)blockIdx.x * 256 + threadIdx.x;      // float4 index
  size_t total = (size_t)B_ * TC * (DD / 4);
  if (idx >= total) return;
  int d4 = (int)(idx & (DD / 4 - 1));
  size_t rowid = idx >> 6;
  int bb = (int)(rowid >> 12);
  float4 c4 = *(const float4*)(ctx + rowid * DD + d4 * 4);
  float4 h4 = *(const float4*)(h + bb * DD + d4 * 4);
  float4 o;
  o.x = c4.x * h4.x; o.y = c4.y * h4.y; o.z = c4.z * h4.z; o.w = c4.w * h4.w;
  *(float4*)(out + rowid * (4 * DD) + 3 * DD + d4 * 4) = o;
}

extern "C" void kernel_launch(void* const* d_in, const int* in_sizes, int n_in,
                              void* d_out, int out_size, void* d_ws, size_t ws_size,
                              hipStream_t stream) {
  const float* ctx = (const float*)d_in[0];
  const float* qst = (const float*)d_in[1];
  const float* w   = (const float*)d_in[2];
  float* out   = (float*)d_out;
  float* h_acc = (float*)d_ws;            // B*D floats
  float* sq_g  = h_acc + B_ * DD;         // B*Tq floats

  (void)hipFuncSetAttribute((const void*)bidaf_main,
                            hipFuncAttributeMaxDynamicSharedMemorySize, LDS_BYTES);

  bidaf_setup<<<(B_ * TQ + 255) / 256, 256, 0, stream>>>(qst, w, h_acc, sq_g);
  bidaf_main<<<B_ * (TC / M_TILE), 256, LDS_BYTES, stream>>>(ctx, qst, w, sq_g, h_acc, out);
  bidaf_tail<<<(int)(((size_t)B_ * TC * (DD / 4) + 255) / 256), 256, 0, stream>>>(ctx, h_acc, out);
}
